// MyModel_83588653515257
// MI455X (gfx1250) — matrix operand order repaired and checked by reading, not compiled, not run on hardware
//
#include <hip/hip_runtime.h>

typedef __attribute__((ext_vector_type(16))) _Float16 v16h;
typedef __attribute__((ext_vector_type(8)))  float    v8f;

#define F 40
#define HDIM 40
#define WDIM 40
#define TDIM 32
#define BDIM 16
#define NGATE 160      // 4F
#define NTILES 10      // 160 / 16
#define BN_EPS 1e-3f

__device__ __forceinline__ float hsig(float x) {
    float y = 0.2f * x + 0.5f;
    return fminf(fmaxf(y, 0.f), 1.f);
}

// ---------------------------------------------------------------------------
// f32 -> f16 input conversion ([B,T,H,W,1])
// ---------------------------------------------------------------------------
__global__ void k_cvt_input(const float* __restrict__ x, _Float16* __restrict__ y, int n) {
    int i = blockIdx.x * 256 + threadIdx.x;
    if (i < n) y[i] = (_Float16)x[i];
}

// ---------------------------------------------------------------------------
// zero h (both buffers) and c state
// ---------------------------------------------------------------------------
__global__ void k_zero_state(_Float16* h0, _Float16* h1, float* c, int n) {
    int i = blockIdx.x * 256 + threadIdx.x;
    if (i < n) { h0[i] = (_Float16)0.f; h1[i] = (_Float16)0.f; c[i] = 0.f; }
}

// ---------------------------------------------------------------------------
// Pack combined [Wx;Wh] (channel-interleaved im2col) into WMMA B-fragment
// order: wp[nt][kc][lane][e],   lane -> N=nt*16+(lane&15),
//                               e    -> K = kc*32 + e + ((lane>>4)<<4)
// ---------------------------------------------------------------------------
__global__ void k_pack_weights(const float* __restrict__ Wx, const float* __restrict__ Wh,
                               _Float16* __restrict__ wp, int cin, int nchunks) {
    int idx = blockIdx.x * 256 + threadIdx.x;
    int total = NTILES * nchunks * 32 * 16;
    if (idx >= total) return;
    int e    = idx & 15;
    int lane = (idx >> 4) & 31;
    int rest = idx >> 9;
    int kc   = rest % nchunks;
    int nt   = rest / nchunks;
    int n    = nt * 16 + (lane & 15);
    int k    = kc * 32 + e + ((lane >> 4) << 4);
    int ctot = cin + F;
    int ktot = 9 * ctot;
    float v = 0.f;
    if (k < ktot) {
        int kk = k / ctot;     // ky*3+kx
        int c  = k % ctot;     // interleaved channel: x then h
        if (c < cin) v = Wx[(kk * cin + c) * NGATE + n];
        else         v = Wh[(kk * F + (c - cin)) * NGATE + n];
    }
    wp[idx] = (_Float16)v;
}

// ---------------------------------------------------------------------------
// One ConvLSTM timestep as implicit GEMM via v_wmma_f32_16x16x32_f16.
// Workgroup: 320 threads = 10 waves. Wave w owns N-tile w (16 gate channels)
// for a 16-position row strip of one batch image.
//
// Data path:
//   1) stage raw im2col patch [3][18][CTOT] (f16) into LDS
//   2) whole-WG repack into afrag[NCH][32][16] in exact WMMA-A fragment
//      order (k->patch decomposition done ONCE, not once per wave)
//   3) per wave: K loop with v16h LDS loads (ds_load_b128 x2) + WMMA,
//      two independent accumulator chains for XDL pipelining
//   4) z exchanged via LDS, gates + state update + fused BN
// ---------------------------------------------------------------------------
template <int CIN>
__global__ void __launch_bounds__(320)
k_convlstm_step(const _Float16* __restrict__ xseq,   // [B,T,H,W,CIN]
                const _Float16* __restrict__ hin,    // [B,H,W,F]
                _Float16*       __restrict__ hout,   // [B,H,W,F]
                float*          __restrict__ cstate, // [B,H,W,F]
                const _Float16* __restrict__ wpack,  // [NTILES][NCH][32][16]
                const float* __restrict__ bias,      // [160]
                const float* __restrict__ gam, const float* __restrict__ bet,
                const float* __restrict__ mu,  const float* __restrict__ var,
                _Float16*       __restrict__ outseq, // [B,T,H,W,F] (BN applied)
                int t) {
    constexpr int CTOT = CIN + F;
    constexpr int KTOT = 9 * CTOT;
    constexpr int NCH  = (KTOT + 31) / 32;
    constexpr int PW   = 18;                 // patch width: w0-1 .. w0+16

    __shared__ _Float16 patch[3 * PW * CTOT];
    __shared__ _Float16 afrag[NCH * 32 * 16];
    __shared__ float    zl[16 * NGATE];

    const int tid  = threadIdx.x;
    const int lane = tid & 31;
    const int wave = tid >> 5;               // 0..9 == N-tile index
    const int wg   = blockIdx.x;
    const int mt   = wg % 120;               // 3 strips per row * 40 rows
    const int b    = wg / 120;
    const int row  = mt / 3;
    const int w0   = (mt % 3) * 16;
    const int mcount = (mt % 3 == 2) ? 8 : 16;

    // ---- 1) stage im2col patch (x channels then h channels, interleaved) --
    const _Float16* xbase = xseq + (size_t)(b * TDIM + t) * (HDIM * WDIM) * CIN;
    const _Float16* hbase = hin + (size_t)b * (HDIM * WDIM) * F;
    for (int i = tid; i < 3 * PW * CTOT; i += 320) {
        int c  = i % CTOT;
        int xi = (i / CTOT) % PW;
        int ky = i / (CTOT * PW);
        int hy = row - 1 + ky;
        int wx = w0 - 1 + xi;
        _Float16 v = (_Float16)0.f;
        if (hy >= 0 && hy < HDIM && wx >= 0 && wx < WDIM) {
            int sp = hy * WDIM + wx;
            v = (c < CIN) ? xbase[sp * CIN + c] : hbase[sp * F + (c - CIN)];
        }
        patch[i] = v;
    }
    __syncthreads();

    // ---- 2) repack into WMMA-A fragment order (done once per WG) ---------
    for (int idx = tid; idx < NCH * 32 * 16; idx += 320) {
        int m = idx & 15;                    // A-matrix row (spatial position)
        int k = idx >> 4;                    // global K index (incl. zero pad)
        _Float16 v = (_Float16)0.f;
        if (k < KTOT) {
            int kk = k / CTOT;               // constexpr divisor -> cheap
            int c  = k % CTOT;
            int ky = kk / 3, kx = kk % 3;
            v = patch[(ky * PW + (m + kx)) * CTOT + c];
        }
        int kc     = k >> 5;
        int klocal = k & 31;
        int half   = klocal >> 4;            // which 16-lane half holds this K
        int alane  = half * 16 + m;
        int e      = klocal & 15;
        afrag[(kc * 32 + alane) * 16 + e] = v;
    }
    __syncthreads();

    // ---- 3) K loop: wide LDS loads + WMMA, dual accumulator chains -------
    v8f acc0 = {}, acc1 = {};
    const v16h* af = (const v16h*)afrag;     // index: kc*32 + lane
    const _Float16* wpb = wpack + ((size_t)(wave * NCH) * 32 + lane) * 16;
    int kc = 0;
    for (; kc + 1 < NCH; kc += 2) {
        __builtin_prefetch(wpb + (size_t)(kc + 2) * 512, 0, 0);
        v16h a0 = af[kc * 32 + lane];
        v16h b0 = *(const v16h*)(wpb + (size_t)kc * 512);
        v16h a1 = af[(kc + 1) * 32 + lane];
        v16h b1 = *(const v16h*)(wpb + (size_t)(kc + 1) * 512);
        acc0 = __builtin_amdgcn_wmma_f32_16x16x32_f16(false, a0, false, b0,
                                                      (short)0, acc0, false, false);
        acc1 = __builtin_amdgcn_wmma_f32_16x16x32_f16(false, a1, false, b1,
                                                      (short)0, acc1, false, false);
    }
    if (kc < NCH) {
        v16h a0 = af[kc * 32 + lane];
        v16h b0 = *(const v16h*)(wpb + (size_t)kc * 512);
        acc0 = __builtin_amdgcn_wmma_f32_16x16x32_f16(false, a0, false, b0,
                                                      (short)0, acc0, false, false);
    }
#pragma unroll
    for (int r = 0; r < 8; ++r) acc0[r] += acc1[r];

    // ---- scatter z to LDS so gates (different N-tiles) can be combined ----
#pragma unroll
    for (int r = 0; r < 8; ++r) {
        int mm = r + ((lane >> 4) << 3);
        int nn = wave * 16 + (lane & 15);
        zl[mm * NGATE + nn] = acc0[r];
    }
    __syncthreads();

    // ---- 4) gates, cell/state update, fused BN, stores -------------------
    float*    cbase  = cstate + (size_t)b * (HDIM * WDIM) * F;
    _Float16* hobase = hout   + (size_t)b * (HDIM * WDIM) * F;
    _Float16* obase  = outseq + (size_t)(b * TDIM + t) * (HDIM * WDIM) * F;
    for (int i = tid; i < mcount * F; i += 320) {
        int fc = i % F;
        int mm = i / F;
        int sp = row * WDIM + (w0 + mm);
        float zi = zl[mm * NGATE + fc]         + bias[fc];
        float zf = zl[mm * NGATE + F + fc]     + bias[F + fc];
        float zg = zl[mm * NGATE + 2 * F + fc] + bias[2 * F + fc];
        float zo = zl[mm * NGATE + 3 * F + fc] + bias[3 * F + fc];
        float cold = cbase[sp * F + fc];
        float cn = hsig(zf) * cold + hsig(zi) * tanhf(zg);
        float hn = hsig(zo) * tanhf(cn);
        cbase[sp * F + fc]  = cn;
        hobase[sp * F + fc] = (_Float16)hn;
        float scale = gam[fc] * rsqrtf(var[fc] + BN_EPS);
        obase[sp * F + fc] = (_Float16)((hn - mu[fc]) * scale + bet[fc]);
    }
}

// ---------------------------------------------------------------------------
// Conv3D (3x3x3, F->1, SAME) + sigmoid. Tiny (~1.8 GFLOP): plain VALU kernel,
// weights cached in LDS.
// ---------------------------------------------------------------------------
__global__ void __launch_bounds__(256)
k_conv3d_sigmoid(const _Float16* __restrict__ seq,  // [B,T,H,W,F]
                 const float* __restrict__ w,       // [3][3][3][F]
                 const float* __restrict__ b3,
                 float* __restrict__ out) {         // [B,T,H,W]
    __shared__ float wl[27 * F];
    for (int i = threadIdx.x; i < 27 * F; i += 256) wl[i] = w[i];
    __syncthreads();
    int idx = blockIdx.x * 256 + threadIdx.x;
    int total = BDIM * TDIM * HDIM * WDIM;
    if (idx >= total) return;
    int wx = idx % WDIM;
    int hy = (idx / WDIM) % HDIM;
    int t  = (idx / (WDIM * HDIM)) % TDIM;
    int b  = idx / (WDIM * HDIM * TDIM);
    float accv = b3[0];
    for (int kt = 0; kt < 3; ++kt) {
        int tt = t - 1 + kt;
        if (tt < 0 || tt >= TDIM) continue;
        for (int ky = 0; ky < 3; ++ky) {
            int yy = hy - 1 + ky;
            if (yy < 0 || yy >= HDIM) continue;
            for (int kx = 0; kx < 3; ++kx) {
                int xx = wx - 1 + kx;
                if (xx < 0 || xx >= WDIM) continue;
                const _Float16* p = seq + (((size_t)(b * TDIM + tt) * HDIM + yy) * WDIM + xx) * F;
                const float* wf = wl + ((kt * 3 + ky) * 3 + kx) * F;
#pragma unroll 8
                for (int c = 0; c < F; ++c) accv += (float)p[c] * wf[c];
            }
        }
    }
    out[idx] = 1.f / (1.f + __expf(-accv));
}

// ---------------------------------------------------------------------------
// Host orchestration: input cvt, then per layer (pack weights, zero state,
// 32 sequential timestep launches), then Conv3D+sigmoid.
// ---------------------------------------------------------------------------
extern "C" void kernel_launch(void* const* d_in, const int* in_sizes, int n_in,
                              void* d_out, int out_size, void* d_ws, size_t ws_size,
                              hipStream_t stream) {
    (void)in_sizes; (void)n_in; (void)out_size; (void)ws_size;
    const float* x   = (const float*)d_in[0];
    const float* W3d = (const float*)d_in[29];
    const float* b3d = (const float*)d_in[30];

    char* ws = (char*)d_ws;
    size_t off = 0;
    auto carve = [&](size_t bytes) -> char* {
        char* p = ws + off;
        off += (bytes + 255) & ~(size_t)255;
        return p;
    };
    const size_t nseq = (size_t)BDIM * TDIM * HDIM * WDIM;  // 819200
    const size_t nst  = (size_t)BDIM * HDIM * WDIM * F;     // 1024000
    _Float16* seqIn = (_Float16*)carve(nseq * 1 * sizeof(_Float16));
    _Float16* p0    = (_Float16*)carve(nseq * F * sizeof(_Float16));
    _Float16* p1    = (_Float16*)carve(nseq * F * sizeof(_Float16));
    _Float16* h0    = (_Float16*)carve(nst * sizeof(_Float16));
    _Float16* h1    = (_Float16*)carve(nst * sizeof(_Float16));
    float*    cst   = (float*)carve(nst * sizeof(float));
    _Float16* wp    = (_Float16*)carve((size_t)23 * NTILES * 32 * 16 * sizeof(_Float16));

    k_cvt_input<<<(int)((nseq + 255) / 256), 256, 0, stream>>>(x, seqIn, (int)nseq);

    _Float16* bufs[5] = {seqIn, p1, p0, p1, p0};
    for (int l = 0; l < 4; ++l) {
        const float* Wx = (const float*)d_in[1 + 7 * l];
        const float* Wh = (const float*)d_in[2 + 7 * l];
        const float* bb = (const float*)d_in[3 + 7 * l];
        const float* gg = (const float*)d_in[4 + 7 * l];
        const float* be = (const float*)d_in[5 + 7 * l];
        const float* mm = (const float*)d_in[6 + 7 * l];
        const float* vv = (const float*)d_in[7 + 7 * l];
        int cin = (l == 0) ? 1 : F;
        int nch = (9 * (cin + F) + 31) / 32;
        int tot = NTILES * nch * 32 * 16;
        k_pack_weights<<<(tot + 255) / 256, 256, 0, stream>>>(Wx, Wh, wp, cin, nch);
        k_zero_state<<<(int)((nst + 255) / 256), 256, 0, stream>>>(h0, h1, cst, (int)nst);
        _Float16* in   = bufs[l];
        _Float16* outb = bufs[l + 1];
        for (int t = 0; t < TDIM; ++t) {
            _Float16* hi = (t & 1) ? h1 : h0;
            _Float16* ho = (t & 1) ? h0 : h1;
            if (l == 0)
                k_convlstm_step<1><<<BDIM * 120, 320, 0, stream>>>(
                    in, hi, ho, cst, wp, bb, gg, be, mm, vv, outb, t);
            else
                k_convlstm_step<F><<<BDIM * 120, 320, 0, stream>>>(
                    in, hi, ho, cst, wp, bb, gg, be, mm, vv, outb, t);
        }
    }
    k_conv3d_sigmoid<<<(int)((nseq + 255) / 256), 256, 0, stream>>>(
        p0, W3d, b3d, (float*)d_out);
}
